// SelfAttention_24137716204000
// MI455X (gfx1250) — matrix operand order repaired and checked by reading, not compiled, not run on hardware
//
#include <hip/hip_runtime.h>

typedef unsigned short u16;
typedef __attribute__((ext_vector_type(16))) __bf16 v16bf;
typedef __attribute__((ext_vector_type(8)))  float  v8f;

#define EMBED_DIM 1024
#define NUM_HEADS 16
#define HEAD_DIM  64
#define SEQ       2048
#define BATCH     2
#define NROWS     (BATCH * SEQ)   // 4096

// CDNA5 async global->LDS copies (GLOBAL_LOAD_ASYNC_TO_LDS_B128, ASYNCcnt).
#define USE_ASYNC_LDS 1

union Frag {
    int4  i[2];
    v16bf v;
};

__device__ __forceinline__ u16 f2bf(float f) {
    unsigned int u = __float_as_uint(f);
    u += 0x7FFFu + ((u >> 16) & 1u);   // round-to-nearest-even
    return (u16)(u >> 16);
}

// global -> LDS b128 copy. Async form avoids the VGPR round-trip and is
// tracked with ASYNCcnt (one s_wait_asynccnt 0 before the consuming barrier).
__device__ __forceinline__ void copy_b128(const u16* g, u16* l) {
#if USE_ASYNC_LDS
    unsigned loff = (unsigned)(size_t)l;              // low 32 bits = LDS address
    unsigned long long ga = (unsigned long long)(size_t)g;
    asm volatile("global_load_async_to_lds_b128 %0, %1, off"
                 :: "v"(loff), "v"(ga) : "memory");
#else
    *(int4*)l = *(const int4*)g;
#endif
}

__device__ __forceinline__ void copy_fence() {
#if USE_ASYNC_LDS
    asm volatile("s_wait_asynccnt 0x0" ::: "memory");
#endif
}

// ---------------------------------------------------------------------------
// fp32 -> bf16 conversion, 4 elements/thread
// ---------------------------------------------------------------------------
__global__ __launch_bounds__(256) void cvt_f32_bf16(const float* __restrict__ in,
                                                    u16* __restrict__ out, int n) {
    int i = (blockIdx.x * blockDim.x + threadIdx.x) * 4;
    if (i + 3 < n) {
        float4 f = *(const float4*)(in + i);
        ushort4 o;
        o.x = f2bf(f.x); o.y = f2bf(f.y); o.z = f2bf(f.z); o.w = f2bf(f.w);
        *(ushort4*)(out + i) = o;
    } else {
        for (int j = i; j < n; ++j) out[j] = f2bf(in[j]);
    }
}

// ---------------------------------------------------------------------------
// fp32 (RxC row-major) -> bf16 transposed (CxR row-major). One-time for weights.
// ---------------------------------------------------------------------------
__global__ __launch_bounds__(256) void cvt_transpose_bf16(const float* __restrict__ in,
                                                          u16* __restrict__ out,
                                                          int R, int C) {
    __shared__ u16 Ts[32][33];
    const int tx = threadIdx.x, ty = threadIdx.y;
    const int c0 = blockIdx.x * 32, r0 = blockIdx.y * 32;
#pragma unroll
    for (int j = 0; j < 4; ++j)
        Ts[ty + j * 8][tx] = f2bf(in[(size_t)(r0 + ty + j * 8) * C + c0 + tx]);
    __syncthreads();
#pragma unroll
    for (int j = 0; j < 4; ++j)
        out[(size_t)(c0 + ty + j * 8) * R + r0 + tx] = Ts[tx][ty + j * 8];
}

// ---------------------------------------------------------------------------
// GEMM (TN): out = (A(MxK,bf16,row) @ Bt(NxK,bf16,row)^T + bias) * alpha
// 128x128 WG tile, 256 threads = 8 waves; wave -> (mw=w&3: 32 rows, nw=w>>2: 64 cols)
// Each wave: 2x4 grid of 16x16 f32 accumulators = 8 WMMAs per 32-K stage.
// Double-buffered async staging: DMA of tile t+1 overlaps WMMA of tile t.
// ---------------------------------------------------------------------------
__global__ __launch_bounds__(256) void gemm_bf16_tn(const u16* __restrict__ A,
                                                    const u16* __restrict__ Bt,
                                                    const float* __restrict__ bias,
                                                    u16* __restrict__ outb,
                                                    float* __restrict__ outf,
                                                    u16* __restrict__ outbt,
                                                    int M, int N, int K, float alpha) {
    __shared__ __align__(16) u16 As[2 * 128 * 48];   // A tiles 128x32, row stride 48
    __shared__ __align__(16) u16 Bs[2 * 128 * 48];   // Bt tiles 128x32 (row=n, col=k)

    const int tid  = threadIdx.x;
    const int wave = tid >> 5;
    const int lane = tid & 31;
    const int hgrp = lane >> 4;
    const int lm   = lane & 15;

    const int m0 = blockIdx.y * 128;
    const int n0 = blockIdx.x * 128;
    const int mw = wave & 3;      // 32-row slab
    const int nw = wave >> 2;     // 64-col slab

    // staging task (fully unrolled, no predication): rows {srow, srow+64}
    const int srow = tid >> 2;    // 0..63
    const int sch  = tid & 3;     // 0..3
    const int soff = srow * 48 + sch * 8;

    v8f acc[2][4];
#pragma unroll
    for (int i = 0; i < 2; ++i)
#pragma unroll
        for (int c = 0; c < 4; ++c) acc[i][c] = v8f{};

    const u16* gA = A  + (size_t)(m0 + srow) * K + sch * 8;
    const u16* gB = Bt + (size_t)(n0 + srow) * K + sch * 8;

    // prologue: issue async copies for tile 0 into buffer 0
    copy_b128(gA,                  As + soff);
    copy_b128(gA + (size_t)64 * K, As + 64 * 48 + soff);
    copy_b128(gB,                  Bs + soff);
    copy_b128(gB + (size_t)64 * K, Bs + 64 * 48 + soff);

    const int steps = K >> 5;
    for (int s = 0; s < steps; ++s) {
        const int buf = s & 1;
        const u16* as = As + buf * (128 * 48);
        const u16* bs = Bs + buf * (128 * 48);

        copy_fence();        // this wave's tile-s copies have landed
        __syncthreads();     // every wave's tile-s copies have landed

        if (s + 1 < steps) { // overlap: DMA tile s+1 into the other buffer
            const int nb = (s + 1) & 1;
            const int kn = (s + 1) << 5;
            u16* asn = As + nb * (128 * 48);
            u16* bsn = Bs + nb * (128 * 48);
            copy_b128(gA + kn,                  asn + soff);
            copy_b128(gA + kn + (size_t)64 * K, asn + 64 * 48 + soff);
            copy_b128(gB + kn,                  bsn + soff);
            copy_b128(gB + kn + (size_t)64 * K, bsn + 64 * 48 + soff);
        }

        // A fragments: row = lm, K = hgrp*16 + t (contiguous)
        Frag aF[2];
#pragma unroll
        for (int i = 0; i < 2; ++i) {
            const u16* ar = as + (mw * 32 + i * 16 + lm) * 48;
            aF[i].i[0] = *(const int4*)(ar + hgrp * 16);
            aF[i].i[1] = *(const int4*)(ar + hgrp * 16 + 8);
        }
        // B fragments: col = lm, K = hgrp*16 + t (contiguous in Bs rows)
        Frag bF[4];
#pragma unroll
        for (int c = 0; c < 4; ++c) {
            const u16* br = bs + (nw * 64 + c * 16 + lm) * 48 + hgrp * 16;
            bF[c].i[0] = *(const int4*)(br);
            bF[c].i[1] = *(const int4*)(br + 8);
        }
#pragma unroll
        for (int i = 0; i < 2; ++i)
#pragma unroll
            for (int c = 0; c < 4; ++c)
                acc[i][c] = __builtin_amdgcn_wmma_f32_16x16x32_bf16(
                    false, aF[i].v, false, bF[c].v, (short)0, acc[i][c], false, false);
    }

    float bv[4];
#pragma unroll
    for (int c = 0; c < 4; ++c) {
        int col = n0 + nw * 64 + c * 16 + lm;
        bv[c] = bias ? bias[col] : 0.0f;
    }
#pragma unroll
    for (int i = 0; i < 2; ++i) {
#pragma unroll
        for (int c = 0; c < 4; ++c) {
            const int col = n0 + nw * 64 + c * 16 + lm;
#pragma unroll
            for (int r = 0; r < 8; ++r) {
                const int row = m0 + mw * 32 + i * 16 + 8 * hgrp + r;
                const float v = (acc[i][c][r] + bv[c]) * alpha;
                if (outf) outf[(size_t)row * N + col] = v;
                if (outb) outb[(size_t)row * N + col] = f2bf(v);
                if (outbt) {
                    // V: [B,S,(H,Hd)] -> [B,H,Hd,S]
                    const int bb = row >> 11;          // row / SEQ
                    const int ss = row & (SEQ - 1);
                    const int hh = col >> 6;           // col / HEAD_DIM
                    const int dd = col & (HEAD_DIM - 1);
                    outbt[(((size_t)bb * NUM_HEADS + hh) * HEAD_DIM + dd) * SEQ + ss] =
                        f2bf(v);
                }
            }
        }
    }
}

// ---------------------------------------------------------------------------
// Flash attention: 1 WG = 4 waves = 64 queries of one (b,h).
// Q pre-scaled by 1/sqrt(Hd). 32 keys per iteration; double-buffered K/V DMA.
// ---------------------------------------------------------------------------
__global__ __launch_bounds__(128) void attn_bf16(const u16* __restrict__ Q,
                                                 const u16* __restrict__ Km,
                                                 const u16* __restrict__ Vtg,
                                                 u16* __restrict__ O) {
    __shared__ __align__(16) u16 Ks[2 * 32 * 72];   // K tiles row-major (32 keys x 64)
    __shared__ __align__(16) u16 Vt[2 * 64 * 48];   // V tiles: Vt[d][k]
    __shared__ __align__(16) u16 Pt[4 * 16 * 48];   // per-wave P tile (16x32)

    const int tid  = threadIdx.x;
    const int wave = tid >> 5;
    const int lane = tid & 31;
    const int hgrp = lane >> 4;
    const int lm   = lane & 15;

    const int qt = blockIdx.x & 31;          // S/64 = 32 query tiles
    const int bh = blockIdx.x >> 5;
    const int b  = bh >> 4;
    const int h  = bh & 15;

    const size_t head  = ((size_t)b * SEQ) * EMBED_DIM + (size_t)h * HEAD_DIM;
    const size_t vhead = (((size_t)b * NUM_HEADS + h) * HEAD_DIM) * SEQ;
    const int q0w = qt * 64 + wave * 16;

    // staging tasks (fully unrolled): K rows {kr, kr+16}, V rows {vd, vd+32}
    const int kr = tid >> 3, kch = tid & 7;  // 0..15, 0..7
    const int vd = tid >> 2, vch = tid & 3;  // 0..31, 0..3
    const int koff = kr * 72 + kch * 8;
    const int voff = vd * 48 + vch * 8;

    const u16* gK = Km  + head  + (size_t)kr * EMBED_DIM + kch * 8;
    const u16* gV = Vtg + vhead + (size_t)vd * SEQ + vch * 8;

    // Q fragments (16x64 = 2 A-frags), loaded once
    Frag aq0, aq1;
    {
        const u16* qrow = Q + head + (size_t)(q0w + lm) * EMBED_DIM;
        aq0.i[0] = *(const int4*)(qrow + hgrp * 16);
        aq0.i[1] = *(const int4*)(qrow + hgrp * 16 + 8);
        aq1.i[0] = *(const int4*)(qrow + 32 + hgrp * 16);
        aq1.i[1] = *(const int4*)(qrow + 40 + hgrp * 16);
    }

    v8f  o[4] = {v8f{}, v8f{}, v8f{}, v8f{}};
    float mr[8], lr[8];
#pragma unroll
    for (int r = 0; r < 8; ++r) { mr[r] = -3.0e38f; lr[r] = 0.0f; }

    // prologue: DMA key-block 0 into buffer 0
    copy_b128(gK,                            Ks + koff);
    copy_b128(gK + (size_t)16 * EMBED_DIM,   Ks + 16 * 72 + koff);
    copy_b128(gV,                            Vt + voff);
    copy_b128(gV + (size_t)32 * SEQ,         Vt + 32 * 48 + voff);

    const int steps = SEQ >> 5;              // 64 key blocks
    for (int s = 0; s < steps; ++s) {
        const int buf = s & 1;
        const u16* ks = Ks + buf * (32 * 72);
        const u16* vt = Vt + buf * (64 * 48);

        copy_fence();
        __syncthreads();

        if (s + 1 < steps) {                 // overlap DMA of next key block
            const int nb = (s + 1) & 1;
            const int kn = (s + 1) << 5;
            u16* ksn = Ks + nb * (32 * 72);
            u16* vtn = Vt + nb * (64 * 48);
            copy_b128(gK + (size_t)kn * EMBED_DIM,        ksn + koff);
            copy_b128(gK + (size_t)(kn + 16) * EMBED_DIM, ksn + 16 * 72 + koff);
            copy_b128(gV + kn,                            vtn + voff);
            copy_b128(gV + kn + (size_t)32 * SEQ,         vtn + 32 * 48 + voff);
        }

        // scores for keys [kb, kb+16) and [kb+16, kb+32)
        v8f s0 = {}, s1 = {};
        {
            Frag bk;
            const u16* kr0 = ks + lm * 72 + hgrp * 16;          // keys 0..15
            bk.i[0] = *(const int4*)(kr0);
            bk.i[1] = *(const int4*)(kr0 + 8);
            s0 = __builtin_amdgcn_wmma_f32_16x16x32_bf16(false, aq0.v, false, bk.v,
                                                         (short)0, s0, false, false);
            bk.i[0] = *(const int4*)(kr0 + 32);
            bk.i[1] = *(const int4*)(kr0 + 40);
            s0 = __builtin_amdgcn_wmma_f32_16x16x32_bf16(false, aq1.v, false, bk.v,
                                                         (short)0, s0, false, false);
            const u16* kr1 = ks + (16 + lm) * 72 + hgrp * 16;   // keys 16..31
            bk.i[0] = *(const int4*)(kr1);
            bk.i[1] = *(const int4*)(kr1 + 8);
            s1 = __builtin_amdgcn_wmma_f32_16x16x32_bf16(false, aq0.v, false, bk.v,
                                                         (short)0, s1, false, false);
            bk.i[0] = *(const int4*)(kr1 + 32);
            bk.i[1] = *(const int4*)(kr1 + 40);
            s1 = __builtin_amdgcn_wmma_f32_16x16x32_bf16(false, aq1.v, false, bk.v,
                                                         (short)0, s1, false, false);
        }

        // online softmax (rows r+8*hgrp live entirely in one 16-lane group)
        float ef[8];
#pragma unroll
        for (int r = 0; r < 8; ++r) {
            float mx = fmaxf(s0[r], s1[r]);
            mx = fmaxf(mx, __shfl_xor(mx, 1, 32));
            mx = fmaxf(mx, __shfl_xor(mx, 2, 32));
            mx = fmaxf(mx, __shfl_xor(mx, 4, 32));
            mx = fmaxf(mx, __shfl_xor(mx, 8, 32));
            float mn = fmaxf(mr[r], mx);
            float p0 = __expf(s0[r] - mn);
            float p1 = __expf(s1[r] - mn);
            float rs = p0 + p1;
            rs += __shfl_xor(rs, 1, 32);
            rs += __shfl_xor(rs, 2, 32);
            rs += __shfl_xor(rs, 4, 32);
            rs += __shfl_xor(rs, 8, 32);
            ef[r] = __expf(mr[r] - mn);
            lr[r] = lr[r] * ef[r] + rs;
            mr[r] = mn;
            // write P (C layout -> LDS) for re-load as A fragment
            u16* prow = Pt + (wave * 16 + 8 * hgrp + r) * 48;
            prow[lm]      = f2bf(p0);
            prow[16 + lm] = f2bf(p1);
        }
        // rescale running output
#pragma unroll
        for (int c = 0; c < 4; ++c)
#pragma unroll
            for (int r = 0; r < 8; ++r) o[c][r] *= ef[r];

        // P as A fragment (16x32) — same-wave LDS ops are in-order
        Frag ap;
        const u16* pr = Pt + (wave * 16 + lm) * 48;
        ap.i[0] = *(const int4*)(pr + hgrp * 16);
        ap.i[1] = *(const int4*)(pr + hgrp * 16 + 8);

        // P @ V : 4 WMMAs over Hd chunks of 16
#pragma unroll
        for (int c = 0; c < 4; ++c) {
            Frag bv;
            const u16* vr = vt + (c * 16 + lm) * 48 + hgrp * 16;
            bv.i[0] = *(const int4*)(vr);
            bv.i[1] = *(const int4*)(vr + 8);
            o[c] = __builtin_amdgcn_wmma_f32_16x16x32_bf16(false, ap.v, false, bv.v,
                                                           (short)0, o[c], false, false);
        }
    }

    // finalize: divide by l and store bf16
#pragma unroll
    for (int r = 0; r < 8; ++r) {
        float inv = 1.0f / lr[r];
        int row = q0w + 8 * hgrp + r;
        u16* orow = O + head + (size_t)row * EMBED_DIM;
#pragma unroll
        for (int c = 0; c < 4; ++c)
            orow[c * 16 + lm] = f2bf(o[c][r] * inv);
    }
}

// ---------------------------------------------------------------------------
// Host-side orchestration
// ---------------------------------------------------------------------------
extern "C" void kernel_launch(void* const* d_in, const int* in_sizes, int n_in,
                              void* d_out, int out_size, void* d_ws, size_t ws_size,
                              hipStream_t stream) {
    const float* x  = (const float*)d_in[0];
    const float* wq = (const float*)d_in[1];
    const float* bq = (const float*)d_in[2];
    const float* wk = (const float*)d_in[3];
    const float* bk = (const float*)d_in[4];
    const float* wv = (const float*)d_in[5];
    const float* bv = (const float*)d_in[6];
    const float* wo = (const float*)d_in[7];
    const float* bo = (const float*)d_in[8];

    const int NX = NROWS * EMBED_DIM;            // 4,194,304
    const int NW = EMBED_DIM * EMBED_DIM;        // 1,048,576

    u16* ws   = (u16*)d_ws;
    u16* Xb   = ws;                // x, bf16 row-major
    u16* Wqt  = Xb  + NX;          // weights, bf16 TRANSPOSED (N x K)
    u16* Wkt  = Wqt + NW;
    u16* Wvt  = Wkt + NW;
    u16* Wot  = Wvt + NW;
    u16* Qb   = Wot + NW;          // Q (pre-scaled), bf16 row-major
    u16* Kb   = Qb  + NX;          // K, bf16 row-major
    u16* Vtg  = Kb  + NX;          // V, bf16 transposed per-head [B][H][Hd][S]
    u16* Ob   = Vtg + NX;          // attention output, bf16 row-major

    cvt_f32_bf16<<<NX / 1024, 256, 0, stream>>>(x, Xb, NX);
    dim3 tblk(32, 8);
    dim3 tgrid(EMBED_DIM / 32, EMBED_DIM / 32);
    cvt_transpose_bf16<<<tgrid, tblk, 0, stream>>>(wq, Wqt, EMBED_DIM, EMBED_DIM);
    cvt_transpose_bf16<<<tgrid, tblk, 0, stream>>>(wk, Wkt, EMBED_DIM, EMBED_DIM);
    cvt_transpose_bf16<<<tgrid, tblk, 0, stream>>>(wv, Wvt, EMBED_DIM, EMBED_DIM);
    cvt_transpose_bf16<<<tgrid, tblk, 0, stream>>>(wo, Wot, EMBED_DIM, EMBED_DIM);

    dim3 ggrid(EMBED_DIM / 128, NROWS / 128);    // (8, 32)
    const float scale = 0.125f;                  // 1/sqrt(64), folded into Q

    gemm_bf16_tn<<<ggrid, 256, 0, stream>>>(Xb, Wqt, bq, Qb, nullptr, nullptr,
                                            NROWS, EMBED_DIM, EMBED_DIM, scale);
    gemm_bf16_tn<<<ggrid, 256, 0, stream>>>(Xb, Wkt, bk, Kb, nullptr, nullptr,
                                            NROWS, EMBED_DIM, EMBED_DIM, 1.0f);
    gemm_bf16_tn<<<ggrid, 256, 0, stream>>>(Xb, Wvt, bv, nullptr, nullptr, Vtg,
                                            NROWS, EMBED_DIM, EMBED_DIM, 1.0f);

    attn_bf16<<<BATCH * NUM_HEADS * (SEQ / 64), 128, 0, stream>>>(Qb, Kb, Vtg, Ob);

    gemm_bf16_tn<<<ggrid, 256, 0, stream>>>(Ob, Wot, bo, nullptr, (float*)d_out,
                                            nullptr, NROWS, EMBED_DIM, EMBED_DIM, 1.0f);
}
